// MultiHeadAttention_14714557956349
// MI455X (gfx1250) — matrix operand order repaired and checked by reading, not compiled, not run on hardware
//
#include <hip/hip_runtime.h>
#include <hip/hip_bf16.h>

#define HIDDEN 1024
#define HEADS  16
#define DHEAD  64
#define NSEQ   2048

typedef __bf16 bf16_t;
typedef __attribute__((ext_vector_type(16))) __bf16 v16bf;
typedef __attribute__((ext_vector_type(8)))  float  v8f;

union Frag { v16bf v; uint4 u[2]; };

static __device__ __forceinline__ v8f wmma_bf16(v16bf a, v16bf b, v8f c) {
  // D = A(16x32 bf16) * B(32x16 bf16) + C(16x16 f32)
  return __builtin_amdgcn_wmma_f32_16x16x32_bf16(false, a, false, b, (short)0, c,
                                                 false, false);
}

// Async copy 16B global -> LDS (CDNA5 GLOBAL_LOAD_ASYNC_TO_LDS_B128, GV mode).
// ldsOff: workgroup-relative LDS byte offset (low 32 bits of flat address).
static __device__ __forceinline__ void async_g2l_b128(unsigned int ldsOff,
                                                      const void* g) {
  asm volatile("global_load_async_to_lds_b128 %0, %1, off"
               :: "v"(ldsOff), "v"((unsigned long long)(uintptr_t)g)
               : "memory");
}
static __device__ __forceinline__ void wait_async_le4() {
  asm volatile("s_wait_asynccnt 4" ::: "memory");
}
static __device__ __forceinline__ void wait_async_le0() {
  asm volatile("s_wait_asynccnt 0" ::: "memory");
}

// ---------------------------------------------------------------------------
// Kernel 1: f32 -> bf16 conversion, 4 elements per lane
// ---------------------------------------------------------------------------
__global__ __launch_bounds__(256) void cvt_f32_to_bf16_x4(
    const float4* __restrict__ src, uint2* __restrict__ dst, int n4) {
  int i = blockIdx.x * blockDim.x + threadIdx.x;
  if (i >= n4) return;
  float4 f = src[i];
  union { bf16_t b[4]; uint2 u; } o;
  o.b[0] = (bf16_t)f.x; o.b[1] = (bf16_t)f.y;
  o.b[2] = (bf16_t)f.z; o.b[3] = (bf16_t)f.w;
  dst[i] = o.u;
}

// ---------------------------------------------------------------------------
// Kernel 2: out = (A @ W^T + bias) * scale      A:[M,1024] bf16, W:[1024,1024]
// One wave computes a 32x64 tile: 2 A-frags x 4 B-frags = 8 WMMAs per K-step.
// out16 != null: bf16 output; transposedOut: store out16[o*M + row] (for V).
// out32 != null: f32 output (final projection).
// ---------------------------------------------------------------------------
__global__ __launch_bounds__(256) void gemm_wt_bf16(
    const bf16_t* __restrict__ A, const bf16_t* __restrict__ W,
    const float* __restrict__ bias, float scale,
    bf16_t* __restrict__ out16, float* __restrict__ out32,
    int transposedOut, int M) {
  const int lane = threadIdx.x & 31;
  const int warp = threadIdx.x >> 5;
  const int hf   = lane >> 4;   // which 16-lane half
  const int ln16 = lane & 15;
  const int colGroups = HIDDEN / 64;           // 16
  const int tile = blockIdx.x * 8 + warp;      // 8 waves / block
  const int row0 = (tile / colGroups) * 32;
  const int col0 = (tile % colGroups) * 64;

  v8f acc[2][4];
  #pragma unroll
  for (int a = 0; a < 2; ++a)
    #pragma unroll
    for (int f = 0; f < 4; ++f)
      #pragma unroll
      for (int r = 0; r < 8; ++r) acc[a][f][r] = 0.0f;

  const bf16_t* aRow0 = A + (size_t)(row0 + ln16) * HIDDEN;
  const bf16_t* aRow1 = A + (size_t)(row0 + 16 + ln16) * HIDDEN;
  for (int k0 = 0; k0 < HIDDEN; k0 += 32) {
    // A fragment: element i -> K = (i/8)*16 + hf*8 + i%8  (two 16B chunks)
    Frag af[2];
    af[0].u[0] = *(const uint4*)(aRow0 + k0 + hf * 8);
    af[0].u[1] = *(const uint4*)(aRow0 + k0 + 16 + hf * 8);
    af[1].u[0] = *(const uint4*)(aRow1 + k0 + hf * 8);
    af[1].u[1] = *(const uint4*)(aRow1 + k0 + 16 + hf * 8);
    #pragma unroll
    for (int f = 0; f < 4; ++f) {
      // B fragment: element i -> K = (i/8)*16 + hf*8 + i%8, N = ln16 (row of W = out col)
      const bf16_t* bPtr =
          W + (size_t)(col0 + f * 16 + ln16) * HIDDEN + k0 + hf * 8;
      Frag bf_;
      bf_.u[0] = *(const uint4*)(bPtr);
      bf_.u[1] = *(const uint4*)(bPtr + 16);
      acc[0][f] = wmma_bf16(af[0].v, bf_.v, acc[0][f]);
      acc[1][f] = wmma_bf16(af[1].v, bf_.v, acc[1][f]);
    }
  }

  #pragma unroll
  for (int a = 0; a < 2; ++a) {
    const int rbase = row0 + a * 16 + hf * 8;
    #pragma unroll
    for (int f = 0; f < 4; ++f) {
      const int o = col0 + f * 16 + ln16;
      const float bval = bias[o];
      if (out32) {
        #pragma unroll
        for (int r = 0; r < 8; ++r)
          out32[(size_t)(rbase + r) * HIDDEN + o] = (acc[a][f][r] + bval) * scale;
      } else if (transposedOut) {
        union { bf16_t b[8]; uint4 u; } pk;
        #pragma unroll
        for (int r = 0; r < 8; ++r)
          pk.b[r] = (bf16_t)((acc[a][f][r] + bval) * scale);
        *(uint4*)(out16 + (size_t)o * M + rbase) = pk.u;
      } else {
        #pragma unroll
        for (int r = 0; r < 8; ++r)
          out16[(size_t)(rbase + r) * HIDDEN + o] =
              (bf16_t)((acc[a][f][r] + bval) * scale);
      }
    }
  }
}

// ---------------------------------------------------------------------------
// Kernel 3: flash attention. One wave per (head, 16-query-row block).
// Per 32-key step: S = Q*K^T (4 wmma) + bias (async-LDS double buffered),
// online softmax, O += P*V (4 wmma).
// ---------------------------------------------------------------------------
__global__ __launch_bounds__(128) void attn_flash_bf16(
    const bf16_t* __restrict__ Qh, const bf16_t* __restrict__ Kh,
    const bf16_t* __restrict__ VhT, const float* __restrict__ abias,
    bf16_t* __restrict__ ctx16) {
  // per-wave P staging (C->A relayout) and double-buffered bias tiles
  __shared__ float pbuf[4][16][32] __attribute__((aligned(16)));
  __shared__ float bbuf[4][2][16][32] __attribute__((aligned(16)));
  const int lane = threadIdx.x & 31;
  const int warp = threadIdx.x >> 5;
  const int hf   = lane >> 4;
  const int ln16 = lane & 15;
  const int wg = blockIdx.x * 4 + warp;     // 2048 waves total
  const int h  = wg >> 7;                   // / (NSEQ/16)
  const int m0 = (wg & 127) * 16;

  // Q tile 16x64 = two A fragments (already pre-scaled by 1/sqrt(d))
  Frag aq0, aq1;
  const bf16_t* qPtr = Qh + (size_t)(m0 + ln16) * HIDDEN + h * DHEAD;
  aq0.u[0] = *(const uint4*)(qPtr + hf * 8);
  aq0.u[1] = *(const uint4*)(qPtr + 16 + hf * 8);
  aq1.u[0] = *(const uint4*)(qPtr + 32 + hf * 8);
  aq1.u[1] = *(const uint4*)(qPtr + 48 + hf * 8);

  v8f oacc[4];
  float rowmax[8], rowsum[8];
  #pragma unroll
  for (int r = 0; r < 8; ++r) { rowmax[r] = -1e30f; rowsum[r] = 0.0f; }
  #pragma unroll
  for (int f = 0; f < 4; ++f)
    #pragma unroll
    for (int r = 0; r < 8; ++r) oacc[f][r] = 0.0f;

  const float* bRow = abias + ((size_t)h * NSEQ + m0) * NSEQ;
  const unsigned int bbufOff[2] = {
      (unsigned int)(uintptr_t)&bbuf[warp][0][0][0],
      (unsigned int)(uintptr_t)&bbuf[warp][1][0][0]};

  // Issue one 16x32 f32 bias tile (2KB) as 4 async B128 copies per lane.
  auto issueBias = [&](int buf, int jcol) {
    #pragma unroll
    for (int i = 0; i < 4; ++i) {
      const int chunk = lane + 32 * i;     // 128 x 16B chunks
      const int row = chunk >> 3;
      const int c4 = (chunk & 7) << 2;     // col (floats)
      async_g2l_b128(bbufOff[buf] + (unsigned int)((row * 32 + c4) * 4),
                     bRow + (size_t)row * NSEQ + jcol + c4);
    }
  };
  issueBias(0, 0);  // preload first tile

  for (int j0 = 0; j0 < NSEQ; j0 += 32) {
    const int buf = (j0 >> 5) & 1;
    const bool more = (j0 + 32) < NSEQ;
    if (more) issueBias(buf ^ 1, j0 + 32);   // prefetch next tile
    if (more) wait_async_le4();              // current tile landed (in-order)
    else      wait_async_le0();

    // ---- scores: two 16x16 tiles over key cols [j0, j0+32) ----
    v8f s[2];
    #pragma unroll
    for (int t = 0; t < 2; ++t) {
      const bf16_t* kPtr =
          Kh + (size_t)(j0 + t * 16 + ln16) * HIDDEN + h * DHEAD;
      Frag bk0, bk1;
      bk0.u[0] = *(const uint4*)(kPtr + hf * 8);
      bk0.u[1] = *(const uint4*)(kPtr + 16 + hf * 8);
      bk1.u[0] = *(const uint4*)(kPtr + 32 + hf * 8);
      bk1.u[1] = *(const uint4*)(kPtr + 48 + hf * 8);
      v8f z = {};
      z = wmma_bf16(aq0.v, bk0.v, z);
      z = wmma_bf16(aq1.v, bk1.v, z);
      #pragma unroll
      for (int r = 0; r < 8; ++r)  // bias tile now in LDS
        z[r] += bbuf[warp][buf][hf * 8 + r][t * 16 + ln16];
      s[t] = z;
    }

    // ---- online softmax (rows split by lane-half, reduce within 16 lanes) --
    float mnew[8], alpha[8];
    #pragma unroll
    for (int r = 0; r < 8; ++r) {
      float tm = fmaxf(s[0][r], s[1][r]);
      #pragma unroll
      for (int off = 8; off >= 1; off >>= 1)
        tm = fmaxf(tm, __shfl_xor(tm, off, 32));
      mnew[r]  = fmaxf(rowmax[r], tm);
      alpha[r] = __expf(rowmax[r] - mnew[r]);
      rowmax[r] = mnew[r];
    }
    #pragma unroll
    for (int r = 0; r < 8; ++r) {
      float p0 = __expf(s[0][r] - mnew[r]);
      float p1 = __expf(s[1][r] - mnew[r]);
      pbuf[warp][hf * 8 + r][ln16]      = p0;
      pbuf[warp][hf * 8 + r][ln16 + 16] = p1;
      float ts = p0 + p1;
      #pragma unroll
      for (int off = 8; off >= 1; off >>= 1) ts += __shfl_xor(ts, off, 32);
      rowsum[r] = rowsum[r] * alpha[r] + ts;
    }
    #pragma unroll
    for (int f = 0; f < 4; ++f)
      #pragma unroll
      for (int r = 0; r < 8; ++r) oacc[f][r] *= alpha[r];

    __syncthreads();
    // ---- P: C-layout (LDS) -> A-layout fragment 16x32, cvt to bf16 ----
    Frag pa;
    {
      const float* prow = &pbuf[warp][ln16][0];
      float tmp[16] __attribute__((aligned(16)));
      *(float4*)&tmp[0]  = *(const float4*)(prow + hf * 8);
      *(float4*)&tmp[4]  = *(const float4*)(prow + hf * 8 + 4);
      *(float4*)&tmp[8]  = *(const float4*)(prow + 16 + hf * 8);
      *(float4*)&tmp[12] = *(const float4*)(prow + 16 + hf * 8 + 4);
      #pragma unroll
      for (int i = 0; i < 16; ++i) pa.v[i] = (bf16_t)tmp[i];
    }
    __syncthreads();

    // ---- O += P * V : V^T layout makes B-fragment loads contiguous ----
    #pragma unroll
    for (int f = 0; f < 4; ++f) {
      const bf16_t* vPtr =
          VhT + (size_t)(h * DHEAD + f * 16 + ln16) * NSEQ + j0 + hf * 8;
      Frag bv_;
      bv_.u[0] = *(const uint4*)(vPtr);
      bv_.u[1] = *(const uint4*)(vPtr + 16);
      oacc[f] = wmma_bf16(pa.v, bv_.v, oacc[f]);
    }
  }

  #pragma unroll
  for (int r = 0; r < 8; ++r) rowsum[r] = 1.0f / rowsum[r];
  bf16_t* cBase = ctx16 + (size_t)(m0 + hf * 8) * HIDDEN + h * DHEAD;
  #pragma unroll
  for (int f = 0; f < 4; ++f)
    #pragma unroll
    for (int r = 0; r < 8; ++r)
      cBase[(size_t)r * HIDDEN + f * 16 + ln16] =
          (bf16_t)(oacc[f][r] * rowsum[r]);
}

// ---------------------------------------------------------------------------
extern "C" void kernel_launch(void* const* d_in, const int* in_sizes, int n_in,
                              void* d_out, int out_size, void* d_ws,
                              size_t ws_size, hipStream_t stream) {
  (void)in_sizes; (void)n_in; (void)out_size; (void)ws_size;
  const float* q  = (const float*)d_in[0];
  const float* k  = (const float*)d_in[1];
  const float* v  = (const float*)d_in[2];
  const float* ab = (const float*)d_in[3];
  const float* Wq = (const float*)d_in[4];
  const float* bq = (const float*)d_in[5];
  const float* Wk = (const float*)d_in[6];
  const float* bk = (const float*)d_in[7];
  const float* Wv = (const float*)d_in[8];
  const float* bv = (const float*)d_in[9];
  const float* Wo = (const float*)d_in[10];
  const float* bo = (const float*)d_in[11];
  float* out = (float*)d_out;

  const size_t NM = (size_t)NSEQ * HIDDEN;    // 2M elems
  const size_t WW = (size_t)HIDDEN * HIDDEN;  // 1M elems
  char* ws = (char*)d_ws;
  size_t off = 0;
  auto alloc16 = [&](size_t elems) -> bf16_t* {
    bf16_t* p = (bf16_t*)(ws + off);
    off += (elems * sizeof(bf16_t) + 255) & ~(size_t)255;
    return p;
  };
  bf16_t* q16   = alloc16(NM);
  bf16_t* k16   = alloc16(NM);
  bf16_t* v16   = alloc16(NM);
  bf16_t* Wq16  = alloc16(WW);
  bf16_t* Wk16  = alloc16(WW);
  bf16_t* Wv16  = alloc16(WW);
  bf16_t* Wo16  = alloc16(WW);
  bf16_t* Qh16  = alloc16(NM);   // Q proj, pre-scaled, [N, H*64]
  bf16_t* Kh16  = alloc16(NM);   // K proj, [N, H*64]
  bf16_t* VhT16 = alloc16(NM);   // V proj transposed, [H*64, N]
  bf16_t* ctx16 = alloc16(NM);   // attention output, [N, H*64]

  auto cvt = [&](const float* s, bf16_t* d, size_t n) {
    int n4 = (int)(n / 4);
    cvt_f32_to_bf16_x4<<<(n4 + 255) / 256, 256, 0, stream>>>(
        (const float4*)s, (uint2*)d, n4);
  };
  cvt(q, q16, NM);   cvt(k, k16, NM);   cvt(v, v16, NM);
  cvt(Wq, Wq16, WW); cvt(Wk, Wk16, WW); cvt(Wv, Wv16, WW); cvt(Wo, Wo16, WW);

  const int gemmBlocks = (NSEQ / 32) * (HIDDEN / 64) / 8;  // 128 blocks
  const float scaleQ = 0.125f;                             // 1/sqrt(64)
  gemm_wt_bf16<<<gemmBlocks, 256, 0, stream>>>(q16, Wq16, bq, scaleQ,
                                               Qh16, nullptr, 0, NSEQ);
  gemm_wt_bf16<<<gemmBlocks, 256, 0, stream>>>(k16, Wk16, bk, 1.0f,
                                               Kh16, nullptr, 0, NSEQ);
  gemm_wt_bf16<<<gemmBlocks, 256, 0, stream>>>(v16, Wv16, bv, 1.0f,
                                               VhT16, nullptr, 1, NSEQ);

  const int attnBlocks = (HEADS * (NSEQ / 16)) / 4;  // 512 blocks, 4 waves ea
  attn_flash_bf16<<<attnBlocks, 128, 0, stream>>>(Qh16, Kh16, VhT16, ab,
                                                  ctx16);

  gemm_wt_bf16<<<gemmBlocks, 256, 0, stream>>>(ctx16, Wo16, bo, 1.0f,
                                               nullptr, out, 0, NSEQ);
}
